// RNNCell_68745246539791
// MI455X (gfx1250) — matrix operand order repaired and checked by reading, not compiled, not run on hardware
//
#include <hip/hip_runtime.h>
#include <hip/hip_bf16.h>
#include <math.h>

typedef __attribute__((ext_vector_type(16))) __bf16 v16bf;
typedef __attribute__((ext_vector_type(8)))  float  v8f;
typedef __attribute__((ext_vector_type(4)))  unsigned int u32x4;
typedef __attribute__((ext_vector_type(8)))  int i32x8;
typedef __attribute__((ext_vector_type(4)))  int i32x4;

#define T_STEPS 8192
#define D_IN    2048
#define D_HID   2048
#define D_OUT   2048

__device__ __forceinline__ float fast_rcp(float x) {
    return __builtin_amdgcn_rcpf(x);          // v_rcp_f32
}
__device__ __forceinline__ float fast_tanh(float x) {
    x = fminf(fmaxf(x, -15.0f), 15.0f);
    float e = __expf(2.0f * x);
    return (e - 1.0f) * fast_rcp(e + 1.0f);
}
__device__ __forceinline__ float fast_sigmoid(float x) {
    return fast_rcp(1.0f + __expf(-x));
}

// ---------------------------------------------------------------------------
// TDM: 2-D bf16 tile load (tile_d1 rows x tile_d0 elems) from a row-major
// tensor (row stride = stride0 elems) into LDS at lds_addr.
// LDS padding: +2 DWORDs (8 B) after every 32 DWORDs (one 64-elem bf16 row)
//   -> LDS row stride = 68 halves (136 B), matching compute-side layout.
// amdgpu-toolchain (clang-23) 6-arg builtin form.
// ---------------------------------------------------------------------------
__device__ __forceinline__
void tdm_load_tile_bf16(unsigned lds_addr, const void* gaddr,
                        unsigned tensor_d0, unsigned tensor_d1,
                        unsigned tile_d0, unsigned tile_d1,
                        unsigned stride0)
{
    unsigned long long ga = (unsigned long long)(size_t)gaddr;
    u32x4 g0;
    g0[0] = 1u;                                   // count=1 (valid), user mode
    g0[1] = lds_addr;                             // LDS byte address
    g0[2] = (unsigned)ga;                         // global_addr[31:0]
    g0[3] = (unsigned)((ga >> 32) & 0x01FFFFFFu)  // global_addr[56:32]
          | (2u << 30);                           // type = 2 ("image")
    i32x8 g1;
    g1[0] = (int)((1u << 16)      // data_size = 1 -> 2-byte elements
                | (1u << 20)      // pad_enable
                | (4u << 22)      // pad_interval: 32 DWORDs (one 64-elem row)
                | (1u << 25));    // pad_amount:   2 DWORDs (8 B)
    g1[1] = (int)((tensor_d0 & 0xFFFFu) << 16);                     // tensor_dim0 lo
    g1[2] = (int)((tensor_d0 >> 16) | ((tensor_d1 & 0xFFFFu) << 16));
    g1[3] = (int)((tensor_d1 >> 16) | ((tile_d0 & 0xFFFFu) << 16)); // tile_dim0
    g1[4] = (int)(tile_d1 & 0xFFFFu);                               // tile_dim1 (tile_dim2=0)
    g1[5] = (int)stride0;                                           // tensor_dim0_stride lo
    g1[6] = 0;
    g1[7] = 0;
    i32x4 z4; z4[0] = 0; z4[1] = 0; z4[2] = 0; z4[3] = 0;
    i32x8 z8;
    #pragma unroll
    for (int i = 0; i < 8; ++i) z8[i] = 0;
    __builtin_amdgcn_tensor_load_to_lds(g0, g1, z4, z4, z8, 0);
}

// ---------------------------------------------------------------------------
// fp32 -> bf16 conversion (grid-stride)
// ---------------------------------------------------------------------------
__global__ __launch_bounds__(256)
void f32_to_bf16_kernel(const float* __restrict__ in, __bf16* __restrict__ out, int n) {
    int i = blockIdx.x * blockDim.x + threadIdx.x;
    int stride = gridDim.x * blockDim.x;
    for (; i < n; i += stride) out[i] = (__bf16)in[i];
}

// ---------------------------------------------------------------------------
// fp32 [R][C] -> bf16 transposed [C][R]  (tiled via LDS)
// ---------------------------------------------------------------------------
__global__ __launch_bounds__(256)
void transpose_f32_to_bf16(const float* __restrict__ in, __bf16* __restrict__ out,
                           int R, int C) {
    __shared__ float tile[32][33];
    const int c0 = blockIdx.x * 32;
    const int r0 = blockIdx.y * 32;
    const int tx = threadIdx.x & 31;
    const int ty = threadIdx.x >> 5;
    #pragma unroll
    for (int i = ty; i < 32; i += 8)
        tile[i][tx] = in[(size_t)(r0 + i) * C + c0 + tx];
    __syncthreads();
    #pragma unroll
    for (int i = ty; i < 32; i += 8)
        out[(size_t)(c0 + i) * R + r0 + tx] = (__bf16)tile[tx][i];
}

// ---------------------------------------------------------------------------
// bf16 WMMA GEMM, TDM-staged + double-buffered:
//   C[M,N] = epi( A[M,K] * Bt[N,K]^T + bias[N] ),  epi: 0=identity, 1=tanh
// Block tile 64x128, 8 waves, each wave owns 32x32 = 2x2 WMMA tiles.
// BK=64: each TDM round feeds 2 k-substeps = 8 WMMAs per barrier window.
// ---------------------------------------------------------------------------
#define BM 64
#define BN 128
#define BK 64
#define LSTRIDE (BK + 4)    // 68 halves = 136 B row stride (TDM pad = 8 B/row)

__global__ __launch_bounds__(256)
void gemm_bf16_f32(const __bf16* __restrict__ A, const __bf16* __restrict__ Bt,
                   const float* __restrict__ bias, float* __restrict__ C,
                   int M, int N, int K, int do_tanh)
{
    __shared__ __align__(16) __bf16 As[2][BM][LSTRIDE];
    __shared__ __align__(16) __bf16 Bs[2][BN][LSTRIDE];

    const int tid  = threadIdx.x;
    const int lane = tid & 31;
    const int wave = tid >> 5;      // 0..7
    const int wr   = wave >> 2;     // 0..1 : 32-row group
    const int wc   = wave & 3;      // 0..3 : 32-col group
    const int m0   = blockIdx.y * BM;
    const int n0   = blockIdx.x * BN;

    v8f acc[2][2];
    #pragma unroll
    for (int sm = 0; sm < 2; ++sm)
        #pragma unroll
        for (int sn = 0; sn < 2; ++sn)
            #pragma unroll
            for (int v = 0; v < 8; ++v) acc[sm][sn][v] = 0.0f;

    const unsigned ldsA[2] = { (unsigned)(size_t)&As[0][0][0],
                               (unsigned)(size_t)&As[1][0][0] };
    const unsigned ldsB[2] = { (unsigned)(size_t)&Bs[0][0][0],
                               (unsigned)(size_t)&Bs[1][0][0] };

    // prologue: DMA tiles for k0 = 0 into buffer 0
    if (wave == 0) {
        tdm_load_tile_bf16(ldsA[0], &A[(size_t)m0 * K],  K, M, BK, BM, K);
        tdm_load_tile_bf16(ldsB[0], &Bt[(size_t)n0 * K], K, N, BK, BN, K);
        __builtin_amdgcn_s_wait_tensorcnt(0);
    }
    __syncthreads();

    int cur = 0;
    for (int k0 = 0; k0 < K; k0 += BK) {
        // kick off next k-step tiles into the other buffer (overlaps WMMA)
        if (wave == 0 && k0 + BK < K) {
            tdm_load_tile_bf16(ldsA[cur ^ 1], &A[(size_t)m0 * K + k0 + BK],
                               K, M, BK, BM, K);
            tdm_load_tile_bf16(ldsB[cur ^ 1], &Bt[(size_t)n0 * K + k0 + BK],
                               K, N, BK, BN, K);
        }

        const int ar = lane & 15;
        const int ak = (lane >> 4) * 16;
        const int bc = lane & 15;
        const int bk = (lane >> 4) * 16;

        #pragma unroll
        for (int kk = 0; kk < 2; ++kk) {        // two 32-wide k substeps
            v16bf afrag[2], bfrag[2];
            #pragma unroll
            for (int sm = 0; sm < 2; ++sm) {
                const __bf16* ap = &As[cur][wr * 32 + sm * 16 + ar][kk * 32];
                #pragma unroll
                for (int i = 0; i < 8; ++i) {
                    afrag[sm][i]     = ap[ak + i];
                    afrag[sm][8 + i] = ap[ak + 8 + i];
                }
            }
            #pragma unroll
            for (int sn = 0; sn < 2; ++sn) {
                const __bf16* bp = &Bs[cur][wc * 32 + sn * 16 + bc][kk * 32 + bk];
                #pragma unroll
                for (int i = 0; i < 16; ++i) bfrag[sn][i] = bp[i];
            }
            #pragma unroll
            for (int sm = 0; sm < 2; ++sm)
                #pragma unroll
                for (int sn = 0; sn < 2; ++sn)
                    acc[sm][sn] = __builtin_amdgcn_wmma_f32_16x16x32_bf16(
                        false, afrag[sm], false, bfrag[sn],
                        (short)0, acc[sm][sn], false, false);
        }

        // make next buffer's DMA visible to all waves
        if (wave == 0) __builtin_amdgcn_s_wait_tensorcnt(0);
        __syncthreads();
        cur ^= 1;
    }

    // epilogue: D layout -> VGPR v: lanes 0-15 row M=v, lanes 16-31 row M=v+8; N=lane%16
    const int cn = lane & 15;
    const int mb = (lane >> 4) * 8;
    #pragma unroll
    for (int sm = 0; sm < 2; ++sm) {
        #pragma unroll
        for (int sn = 0; sn < 2; ++sn) {
            const int gn = n0 + wc * 32 + sn * 16 + cn;
            const float bv = bias[gn];
            #pragma unroll
            for (int v = 0; v < 8; ++v) {
                const int gm = m0 + wr * 32 + sm * 16 + mb + v;
                float val = acc[sm][sn][v] + bv;
                if (do_tanh) val = fast_tanh(val);
                C[(size_t)gm * N + gn] = val;
            }
        }
    }
}

// ---------------------------------------------------------------------------
// device-wide sense barrier (sequential scan)
// ---------------------------------------------------------------------------
__device__ __forceinline__
void grid_barrier(unsigned* cnt, unsigned* gen, unsigned target, unsigned nwg) {
    __syncthreads();
    if (threadIdx.x == 0) {
        __threadfence();
        unsigned prev = __hip_atomic_fetch_add(cnt, 1u, __ATOMIC_ACQ_REL,
                                               __HIP_MEMORY_SCOPE_AGENT);
        if (prev == nwg - 1u) {
            __hip_atomic_store(cnt, 0u, __ATOMIC_RELAXED, __HIP_MEMORY_SCOPE_AGENT);
            __hip_atomic_fetch_add(gen, 1u, __ATOMIC_RELEASE, __HIP_MEMORY_SCOPE_AGENT);
        } else {
            while (__hip_atomic_load(gen, __ATOMIC_ACQUIRE,
                                     __HIP_MEMORY_SCOPE_AGENT) < target) {
                __builtin_amdgcn_s_sleep(2);
            }
        }
    }
    __syncthreads();
}

// ---------------------------------------------------------------------------
// persistent recurrence kernel: 64 WGs, each owns 32 columns of W_h
// (2048x32 fp32 = 256 KB) held in LDS for the whole scan.
// dynamic LDS = 2048*32 + 2048 + 256 floats = 271,360 B (<= 320 KB/WGP)
// ---------------------------------------------------------------------------
__global__ __launch_bounds__(256)
void rnn_scan_kernel(const float* __restrict__ Wh, const float* __restrict__ pre,
                     const float* __restrict__ h0, float* __restrict__ hping,
                     __bf16* __restrict__ Hb, float* __restrict__ hfinal,
                     unsigned* __restrict__ bar, int T)
{
    extern __shared__ float smem[];
    float* Ws  = smem;                 // [2048][32]
    float* hs  = smem + D_HID * 32;    // [2048]
    float* red = hs + D_HID;           // [8][32]

    unsigned* cnt = bar;
    unsigned* gen = bar + 1;

    const int tid = threadIdx.x;
    const int c   = tid & 31;
    const int kg  = tid >> 5;
    const int n0  = blockIdx.x * 32;

    for (int r = kg; r < D_HID; r += 8)
        Ws[r * 32 + c] = Wh[(size_t)r * D_HID + n0 + c];
    for (int i = tid; i < D_HID; i += 256) hs[i] = h0[i];
    __syncthreads();

    for (int t = 0; t < T; ++t) {
        float* hw = hping + (size_t)(t & 1) * D_HID;

        float acc = 0.0f;
        const int kbase = kg * 256;
        #pragma unroll 8
        for (int kk = 0; kk < 256; ++kk)
            acc = fmaf(hs[kbase + kk], Ws[(kbase + kk) * 32 + c], acc);
        red[kg * 32 + c] = acc;
        __syncthreads();

        if (tid < 32) {
            float s = red[tid];
            #pragma unroll
            for (int j = 1; j < 8; ++j) s += red[j * 32 + tid];
            s += pre[(size_t)t * D_HID + n0 + tid];
            float h = fast_sigmoid(s);
            hw[n0 + tid] = h;
            Hb[(size_t)t * D_HID + n0 + tid] = (__bf16)h;
            if (t == T - 1) hfinal[n0 + tid] = h;
            if (t + 1 < T)
                __builtin_prefetch(&pre[(size_t)(t + 1) * D_HID + n0 + tid], 0, 0);
        }

        grid_barrier(cnt, gen, (unsigned)(t + 1), gridDim.x);

        if (t + 1 < T) {
            for (int i = tid; i < D_HID; i += 256) hs[i] = hw[i];
            __syncthreads();
        }
    }
}

// ---------------------------------------------------------------------------
// host launcher
// ---------------------------------------------------------------------------
extern "C" void kernel_launch(void* const* d_in, const int* in_sizes, int n_in,
                              void* d_out, int out_size, void* d_ws, size_t ws_size,
                              hipStream_t stream) {
    (void)in_sizes; (void)n_in; (void)out_size; (void)ws_size;

    const float* x  = (const float*)d_in[0];
    const float* h0 = (const float*)d_in[1];
    const float* Wh = (const float*)d_in[2];
    const float* Wi = (const float*)d_in[3];
    const float* Wo = (const float*)d_in[4];
    const float* bs = (const float*)d_in[5];
    const float* bo = (const float*)d_in[6];

    float* out    = (float*)d_out;
    float* hfinal = out + (size_t)T_STEPS * D_OUT;

    char* ws = (char*)d_ws;
    size_t off = 0;
    __bf16* xb   = (__bf16*)(ws + off); off += (size_t)T_STEPS * D_IN  * 2;
    __bf16* wibT = (__bf16*)(ws + off); off += (size_t)D_IN   * D_HID * 2;
    __bf16* wobT = (__bf16*)(ws + off); off += (size_t)D_HID  * D_OUT * 2;
    float*  pre  = (float*) (ws + off); off += (size_t)T_STEPS * D_HID * 4;
    __bf16* Hb   = (__bf16*)(ws + off); off += (size_t)T_STEPS * D_HID * 2;
    float*  hp   = (float*) (ws + off); off += 2 * (size_t)D_HID * 4;
    unsigned* bar = (unsigned*)(ws + off); off += 256;

    (void)hipMemsetAsync(bar, 0, 256, stream);

    f32_to_bf16_kernel<<<2048, 256, 0, stream>>>(x, xb, T_STEPS * D_IN);
    transpose_f32_to_bf16<<<dim3(D_HID / 32, D_IN / 32), 256, 0, stream>>>(
        Wi, wibT, D_IN, D_HID);
    transpose_f32_to_bf16<<<dim3(D_OUT / 32, D_HID / 32), 256, 0, stream>>>(
        Wo, wobT, D_HID, D_OUT);

    gemm_bf16_f32<<<dim3(D_HID / BN, T_STEPS / BM), 256, 0, stream>>>(
        xb, wibT, bs, pre, T_STEPS, D_HID, D_IN, /*tanh=*/0);

    size_t shmem = (size_t)(D_HID * 32 + D_HID + 256) * sizeof(float);
    (void)hipFuncSetAttribute((const void*)rnn_scan_kernel,
                              hipFuncAttributeMaxDynamicSharedMemorySize, (int)shmem);
    rnn_scan_kernel<<<D_HID / 32, 256, shmem, stream>>>(
        Wh, pre, h0, hp, Hb, hfinal, bar, T_STEPS);

    gemm_bf16_f32<<<dim3(D_OUT / BN, T_STEPS / BM), 256, 0, stream>>>(
        Hb, wobT, bo, out, T_STEPS, D_OUT, D_HID, /*tanh=*/1);
}
